// FullAttention_3994319585449
// MI455X (gfx1250) — compile-verified, matrix operand order checked by reading, not run on hardware
//
#include <hip/hip_runtime.h>

typedef __attribute__((ext_vector_type(16))) __bf16 v16bf;
typedef __attribute__((ext_vector_type(8)))  __bf16 v8bf;
typedef __attribute__((ext_vector_type(4)))  __bf16 v4bf;
typedef __attribute__((ext_vector_type(8)))  float  v8f;
typedef int v4i_vec __attribute__((vector_size(16)));   // matches builtin param type

#define NH 8
#define DH 64
#define NB 4
#define NS 2048
#define NF 512

// ---- CDNA5 async global->LDS path (gated; sync fallback keeps compile green)
#if defined(__has_builtin)
#if __has_builtin(__builtin_amdgcn_global_load_async_to_lds_b128) && \
    __has_builtin(__builtin_amdgcn_s_wait_asynccnt)
#define USE_ASYNC_LDS 1
#endif
#endif

static __device__ __forceinline__ void cp16_g2l(const __bf16* g, __bf16* l) {
#ifdef USE_ASYNC_LDS
  __builtin_amdgcn_global_load_async_to_lds_b128(
      (__attribute__((address_space(1))) v4i_vec*)g,
      (__attribute__((address_space(3))) v4i_vec*)l, 0, 0);
#else
  *(v8bf*)l = *(const v8bf*)g;
#endif
}
static __device__ __forceinline__ void async_wait0() {
#ifdef USE_ASYNC_LDS
  __builtin_amdgcn_s_wait_asynccnt(0);
#endif
}

// Load a 16-bit A/B WMMA fragment (ISA layout: per lane, 8 bf16 at K=g*8.. and
// 8 bf16 at K=16+g*8..) from two 16-byte aligned addresses.
static __device__ __forceinline__ v16bf frag_ld(const __bf16* p0, const __bf16* p1) {
  v8bf lo = *(const v8bf*)p0;
  v8bf hi = *(const v8bf*)p1;
  return __builtin_shufflevector(lo, hi, 0,1,2,3,4,5,6,7,8,9,10,11,12,13,14,15);
}

static __device__ __forceinline__ v8f wmma_bf16(v16bf a, v16bf b, v8f c) {
  return __builtin_amdgcn_wmma_f32_16x16x32_bf16(false, a, false, b, (short)0, c, false, false);
}

// ---------------------------------------------------------------------------
// Kernel 1: projection GEMM  C[m,n] = sum_k X[m,k] * W[k,n]   (M=8192,N=K=512)
// 128 threads = 4 waves; WG tile 64x64; each wave a 32x32 block (2x2 WMMA).
// Tile staging: batched float4 loads -> registers, then packed bf16 LDS stores
// (avoids the serialized load/wait/cvt/store chains seen in prior codegen).
// transposed==0: out row-major bf16 [8192][512]
// transposed==1: out = vpT bf16 [B][H][DH][S]  (for weights@V B-operand)
// ---------------------------------------------------------------------------
__global__ __launch_bounds__(128) void proj_gemm(
    const float* __restrict__ X, const float* __restrict__ W,
    __bf16* __restrict__ out, int transposed)
{
  __shared__ __bf16 As[64][32];   // [m][k]
  __shared__ __bf16 Bs[64][32];   // [n][k]  (W transposed)
  const int tid  = threadIdx.x;
  const int lane = tid & 31;
  const int wv   = tid >> 5;
  const int l15  = lane & 15;
  const int g    = lane >> 4;
  const int m0   = blockIdx.x * 64;
  const int n0   = blockIdx.y * 64;
  const int wm   = (wv & 1) * 32;
  const int wn   = (wv >> 1) * 32;

  v8f c[2][2] = {};

  for (int k0 = 0; k0 < NF; k0 += 32) {
    // ---- batched loads: 4 float4 of X + 4 float4 of W per thread ----------
    float4 xa[4], wa4[4];
    #pragma unroll
    for (int i = 0; i < 4; ++i) {                // A: 512 float4s, coalesced
      int e = tid + i * 128;
      int m = e >> 3, k4 = (e & 7) * 4;
      xa[i] = *(const float4*)&X[(size_t)(m0 + m) * NF + k0 + k4];
    }
    #pragma unroll
    for (int i = 0; i < 4; ++i) {                // W: 512 float4s, coalesced
      int e = tid + i * 128;
      int k = e >> 4, n4 = (e & 15) * 4;
      wa4[i] = *(const float4*)&W[(size_t)(k0 + k) * NF + n0 + n4];
    }
    // ---- convert + store to LDS -------------------------------------------
    #pragma unroll
    for (int i = 0; i < 4; ++i) {                // A tile: packed b64 stores
      int e = tid + i * 128;
      int m = e >> 3, k4 = (e & 7) * 4;
      v4bf y;
      y[0] = (__bf16)xa[i].x; y[1] = (__bf16)xa[i].y;
      y[2] = (__bf16)xa[i].z; y[3] = (__bf16)xa[i].w;
      *(v4bf*)&As[m][k4] = y;
    }
    #pragma unroll
    for (int i = 0; i < 4; ++i) {                // W tile: transposed stores
      int e = tid + i * 128;
      int k = e >> 4, n4 = (e & 15) * 4;
      Bs[n4 + 0][k] = (__bf16)wa4[i].x;
      Bs[n4 + 1][k] = (__bf16)wa4[i].y;
      Bs[n4 + 2][k] = (__bf16)wa4[i].z;
      Bs[n4 + 3][k] = (__bf16)wa4[i].w;
    }
    if (k0 + 32 < NF) {                          // gfx1250 global_prefetch_b8
      __builtin_prefetch(&X[(size_t)(m0 + (tid >> 1)) * NF + k0 + 32], 0, 0);
      __builtin_prefetch(&W[(size_t)(k0 + 32 + (tid >> 3)) * NF + n0], 0, 0);
    }
    __syncthreads();

    v16bf a[2], b[2];
    #pragma unroll
    for (int mi = 0; mi < 2; ++mi) {
      int r = wm + mi * 16 + l15;
      a[mi] = frag_ld(&As[r][g * 8], &As[r][16 + g * 8]);
    }
    #pragma unroll
    for (int ni = 0; ni < 2; ++ni) {
      int r = wn + ni * 16 + l15;
      b[ni] = frag_ld(&Bs[r][g * 8], &Bs[r][16 + g * 8]);
    }
    #pragma unroll
    for (int mi = 0; mi < 2; ++mi)
      #pragma unroll
      for (int ni = 0; ni < 2; ++ni)
        c[mi][ni] = wmma_bf16(a[mi], b[ni], c[mi][ni]);
    __syncthreads();
  }

  #pragma unroll
  for (int mi = 0; mi < 2; ++mi)
  #pragma unroll
  for (int ni = 0; ni < 2; ++ni)
  #pragma unroll
  for (int i = 0; i < 8; ++i) {
    int m = m0 + wm + mi * 16 + g * 8 + i;       // C layout: row = g*8+i
    int n = n0 + wn + ni * 16 + l15;             //           col = l15
    __bf16 v = (__bf16)c[mi][ni][i];
    if (!transposed) {
      out[(size_t)m * NF + n] = v;
    } else {
      int b2 = m >> 11, s = m & (NS - 1);
      int h  = n >> 6,  d = n & (DH - 1);
      out[(((size_t)b2 * NH + h) * DH + d) * NS + s] = v;
    }
  }
}

// ---------------------------------------------------------------------------
// Kernel 2: online softmax stats (row max m, row sum-of-exp l) per (b,h,q).
// 256 threads = 8 waves; wave h handles head h for 16 q-rows.
// K tiles (32 rows x full 512 features) are staged cooperatively into LDS with
// double-buffered async global->LDS copies; all 8 head-waves fragment-load
// from the shared stage. Row reductions stay in registers via shfl_xor.
// ---------------------------------------------------------------------------
__global__ __launch_bounds__(256) void attn_stats(
    const __bf16* __restrict__ qp, const __bf16* __restrict__ kp,
    float* __restrict__ mOut, float* __restrict__ lOut)
{
  __shared__ __bf16 Ks[2][32][NF];               // 64 KB, double buffered
  const int tid  = threadIdx.x;
  const int lane = tid & 31;
  const int h    = tid >> 5;
  const int l15  = lane & 15;
  const int g    = lane >> 4;
  const int q0   = blockIdx.x * 16;
  const int b    = blockIdx.y;

  const __bf16* qb  = qp + (size_t)b * NS * NF + (size_t)h * DH;
  const __bf16* kbA = kp + (size_t)b * NS * NF;  // all 512 features

  v16bf qa[2];
  #pragma unroll
  for (int dblk = 0; dblk < 2; ++dblk) {
    const __bf16* r = qb + (size_t)(q0 + l15) * NF + dblk * 32;
    qa[dblk] = frag_ld(r + g * 8, r + 16 + g * 8);
  }

  float mrow[8], lrow[8];
  #pragma unroll
  for (int i = 0; i < 8; ++i) { mrow[i] = -1e30f; lrow[i] = 0.f; }

  // stage chunk 0 into buffer 0 (2048 x 16B chunks, fully coalesced)
  #pragma unroll
  for (int i = 0; i < 8; ++i) {
    int e = tid + i * 256, r = e >> 6, c = e & 63;
    cp16_g2l(kbA + (size_t)r * NF + c * 8, &Ks[0][r][c * 8]);
  }
  async_wait0();
  __syncthreads();

  for (int k0 = 0; k0 < NS; k0 += 32) {
    const int bf = (k0 >> 5) & 1;
    if (k0 + 32 < NS) {                          // prefetch next chunk (async)
      #pragma unroll
      for (int i = 0; i < 8; ++i) {
        int e = tid + i * 256, r = e >> 6, c = e & 63;
        cp16_g2l(kbA + (size_t)(k0 + 32 + r) * NF + c * 8, &Ks[bf ^ 1][r][c * 8]);
      }
    }
    #pragma unroll
    for (int t = 0; t < 2; ++t) {
      const __bf16* kr = &Ks[bf][t * 16 + l15][h * DH];
      v16bf ka0 = frag_ld(kr + g * 8,      kr + 16 + g * 8);
      v16bf ka1 = frag_ld(kr + 32 + g * 8, kr + 48 + g * 8);
      v8f s = {};
      s = wmma_bf16(qa[0], ka0, s);
      s = wmma_bf16(qa[1], ka1, s);
      #pragma unroll
      for (int i = 0; i < 8; ++i) {
        float v  = s[i] * 0.125f;                // 1/sqrt(dh), dh=64
        float tm = v;
        #pragma unroll
        for (int msk = 1; msk < 16; msk <<= 1) tm = fmaxf(tm, __shfl_xor(tm, msk, 32));
        float mn = fmaxf(mrow[i], tm);
        float p  = __expf(v - mn);
        #pragma unroll
        for (int msk = 1; msk < 16; msk <<= 1) p += __shfl_xor(p, msk, 32);
        lrow[i] = lrow[i] * __expf(mrow[i] - mn) + p;
        mrow[i] = mn;
      }
    }
    async_wait0();
    __syncthreads();
  }
  if (l15 == 0) {
    size_t base = (((size_t)b * NH + h) * NS) + q0 + g * 8;
    #pragma unroll
    for (int i = 0; i < 8; ++i) { mOut[base + i] = mrow[i]; lOut[base + i] = lrow[i]; }
  }
}

// ---------------------------------------------------------------------------
// Kernel 3: weights = exp(s/8 - m)/l ; mean over heads -> weights_mean ;
// out += weights @ V.  8 waves = 8 heads cooperate per 16 q-rows so the
// cross-head mean is a plain LDS sum (no atomics). K chunk staged via async
// copies; LDS weights tile is re-read in A-fragment layout for output WMMAs
// against transposed V.
// ---------------------------------------------------------------------------
__global__ __launch_bounds__(256) void attn_out(
    const __bf16* __restrict__ qp, const __bf16* __restrict__ kp,
    const __bf16* __restrict__ vpT,
    const float* __restrict__ mIn, const float* __restrict__ lIn,
    float* __restrict__ out, float* __restrict__ meanOut)
{
  __shared__ float  wts[NH][16][32];             // 16 KB
  __shared__ __bf16 Ks[32][NF];                  // 32 KB staged K chunk
  const int tid  = threadIdx.x;
  const int lane = tid & 31;
  const int h    = tid >> 5;
  const int l15  = lane & 15;
  const int g    = lane >> 4;
  const int q0   = blockIdx.x * 16;
  const int b    = blockIdx.y;

  const __bf16* qb  = qp  + (size_t)b * NS * NF + (size_t)h * DH;
  const __bf16* kbA = kp  + (size_t)b * NS * NF;
  const __bf16* vb  = vpT + ((size_t)b * NH + h) * DH * NS;

  v16bf qa[2];
  #pragma unroll
  for (int dblk = 0; dblk < 2; ++dblk) {
    const __bf16* r = qb + (size_t)(q0 + l15) * NF + dblk * 32;
    qa[dblk] = frag_ld(r + g * 8, r + 16 + g * 8);
  }

  float mrow[8], linv[8];
  {
    size_t base = (((size_t)b * NH + h) * NS) + q0 + g * 8;
    #pragma unroll
    for (int i = 0; i < 8; ++i) { mrow[i] = mIn[base + i]; linv[i] = 1.f / lIn[base + i]; }
  }

  v8f co[4] = {};

  for (int k0 = 0; k0 < NS; k0 += 32) {
    #pragma unroll
    for (int i = 0; i < 8; ++i) {                // stage K chunk (coalesced)
      int e = tid + i * 256, r = e >> 6, c = e & 63;
      cp16_g2l(kbA + (size_t)(k0 + r) * NF + c * 8, &Ks[r][c * 8]);
    }
    async_wait0();
    __syncthreads();                             // B1: Ks visible

    #pragma unroll
    for (int t = 0; t < 2; ++t) {                // two 16x16 score tiles
      const __bf16* kr = &Ks[t * 16 + l15][h * DH];
      v16bf ka0 = frag_ld(kr + g * 8,      kr + 16 + g * 8);
      v16bf ka1 = frag_ld(kr + 32 + g * 8, kr + 48 + g * 8);
      v8f s = {};
      s = wmma_bf16(qa[0], ka0, s);
      s = wmma_bf16(qa[1], ka1, s);
      #pragma unroll
      for (int i = 0; i < 8; ++i) {
        float w = __expf(s[i] * 0.125f - mrow[i]) * linv[i];
        wts[h][g * 8 + i][t * 16 + l15] = w;
      }
    }
    __syncthreads();                             // B2: wts visible

    // cross-head mean: 512 positions, 256 threads x 2; NT store (64MB stream)
    #pragma unroll
    for (int rep = 0; rep < 2; ++rep) {
      int p  = tid + rep * 256;
      int ql = p >> 5, kl = p & 31;
      float sum = 0.f;
      #pragma unroll
      for (int hh = 0; hh < NH; ++hh) sum += wts[hh][ql][kl];
      __builtin_nontemporal_store(sum * 0.125f,
          &meanOut[((size_t)b * NS + q0 + ql) * NS + k0 + kl]);
    }

    // re-read weights in A-fragment layout, convert to bf16
    const float* wr = &wts[h][l15][0];
    v16bf wa;
    #pragma unroll
    for (int j = 0; j < 8; ++j) {
      wa[j]     = (__bf16)wr[g * 8 + j];
      wa[8 + j] = (__bf16)wr[16 + g * 8 + j];
    }

    #pragma unroll
    for (int j = 0; j < 4; ++j) {                // out tiles over dh blocks
      const __bf16* vr = vb + (size_t)(j * 16 + l15) * NS + k0;
      v16bf vfrag = frag_ld(vr + g * 8, vr + 16 + g * 8);
      co[j] = wmma_bf16(wa, vfrag, co[j]);
    }
    __syncthreads();                             // B3: wts/Ks free for reuse
  }

  #pragma unroll
  for (int j = 0; j < 4; ++j)
  #pragma unroll
  for (int i = 0; i < 8; ++i) {
    int q = q0 + g * 8 + i;
    int n = h * DH + j * 16 + l15;
    out[((size_t)b * NS + q) * NF + n] = co[j][i];
  }
}

// ---------------------------------------------------------------------------
extern "C" void kernel_launch(void* const* d_in, const int* in_sizes, int n_in,
                              void* d_out, int out_size, void* d_ws, size_t ws_size,
                              hipStream_t stream) {
  (void)in_sizes; (void)n_in; (void)out_size; (void)ws_size;
  const float* q  = (const float*)d_in[0];
  const float* k  = (const float*)d_in[1];
  const float* v  = (const float*)d_in[2];
  const float* Wq = (const float*)d_in[3];
  const float* Wk = (const float*)d_in[4];
  const float* Wv = (const float*)d_in[5];

  const size_t nProj = (size_t)NB * NS * NF;     // 4,194,304 elements
  __bf16* qp  = (__bf16*)d_ws;
  __bf16* kp  = qp + nProj;
  __bf16* vpT = kp + nProj;
  float*  mS  = (float*)(vpT + nProj);
  float*  lS  = mS + (size_t)NB * NH * NS;

  float* outp  = (float*)d_out;                  // [B,S,F]
  float* meanp = outp + nProj;                   // [B,S,S]

  dim3 gProj(NB * NS / 64, NF / 64);             // 128 x 8
  proj_gemm<<<gProj, 128, 0, stream>>>(q, Wq, qp, 0);
  proj_gemm<<<gProj, 128, 0, stream>>>(k, Wk, kp, 0);
  proj_gemm<<<gProj, 128, 0, stream>>>(v, Wv, vpT, 1);

  dim3 gAttn(NS / 16, NB);                       // 128 x 4
  attn_stats<<<gAttn, 256, 0, stream>>>(qp, kp, mS, lS);
  attn_out<<<gAttn, 256, 0, stream>>>(qp, kp, vpT, mS, lS, outp, meanp);
}
